// MultiHeadRichAttention_49039936586047
// MI455X (gfx1250) — matrix operand order repaired and checked by reading, not compiled, not run on hardware
//
#include <hip/hip_runtime.h>

// ---------------- problem constants ----------------
#define BB   2048
#define SS   200
#define DD   64
#define HH   4
#define H1C  64
#define H2C  32
#define FF   256        // 4*D
#define SP   208        // S padded to 13 tiles of 16
#define MT   13         // M tiles
#define STX  264        // row stride (bf16 elems) for x / h1 in LDS (= 256 + 8, keeps 16B align)

typedef __attribute__((ext_vector_type(16))) __bf16 bf16x16;
typedef __attribute__((ext_vector_type(8)))  __bf16 bf16x8;
typedef __attribute__((ext_vector_type(8)))  float  f32x8;

// ---------------- LDS layout (bytes) ----------------
#define OFF_X      0
#define OFF_H1     109824                 // SP*STX*2
#define OFF_KEYS   219648                 // + SP*STX*2
#define OFF_SCORES 270848                 // + S*D*4 (51200)
#define OFF_WBAR   274176                 // + H*SP*4 (3328)
#define OFF_QF     275008                 // + SP*4 (832)
#define OFF_COMB   275264                 // + D*4
#define SMEM_BYTES 275520                 // + D*4

// =====================================================================
// Prep: convert W1/W2 (fp32) into bf16 WMMA-B fragments in workspace.
// B fragment layout for v_wmma_*_16x16x32 (bf16), per ISA:
//   lane<16 : N = lane,     elements e=0..7 hold K = kstep*32 + e, e=8..15 hold K = kstep*32 + 8 + e
//   lane>=16: N = lane-16,  elements e=0..7 hold K = kstep*32 + 8 + e, e=8..15 hold K = kstep*32 + 16 + e
// B1pack index: (((nt*8 + kk)*32 + lane)*16 + e)         nt in [0,16)
// B2pack index: ((((h*2+nt2)*2+kk2)*32 + lane)*16 + e)
// =====================================================================
__global__ __launch_bounds__(256) void pack_weights_kernel(
    const float* __restrict__ W1, const float* __restrict__ W2,
    __bf16* __restrict__ B1pack, __bf16* __restrict__ B2pack)
{
  int i = blockIdx.x * 256 + threadIdx.x;
  if (i < 16 * 8 * 32 * 16) {                  // 65536 W1 fragment elems
    int e    = i & 15;
    int lane = (i >> 4) & 31;
    int kk   = (i >> 9) & 7;
    int nt   = i >> 12;
    int Ng   = nt * 16 + (lane & 15);          // global N column (head-major)
    int K    = kk * 32 + ((lane >> 4) * 8) + (e & 7) + ((e >> 3) * 16);
    int h    = Ng >> 6;
    int n    = Ng & 63;
    B1pack[i] = (__bf16)W1[(h * FF + K) * H1C + n];
  } else if (i < 65536 + HH * 2 * 2 * 32 * 16) {
    int j    = i - 65536;
    int e    = j & 15;
    int lane = (j >> 4) & 31;
    int kk2  = (j >> 9) & 1;
    int nt2  = (j >> 10) & 1;
    int h    = j >> 11;
    int N    = nt2 * 16 + (lane & 15);
    int K    = kk2 * 32 + ((lane >> 4) * 8) + (e & 7) + ((e >> 3) * 16);
    B2pack[j] = (__bf16)W2[(h * H1C + K) * H2C + N];
  }
}

// =====================================================================
// Fused kernel: one workgroup (8 wave32) per batch row.
// =====================================================================
__global__ __launch_bounds__(256) void rich_attn_kernel(
    const float* __restrict__ query, const float* __restrict__ keys,
    const int*   __restrict__ keys_mask,
    const __bf16* __restrict__ B1pack, const __bf16* __restrict__ B2pack,
    const float* __restrict__ b1, const float* __restrict__ a1,
    const float* __restrict__ b2, const float* __restrict__ a2,
    const float* __restrict__ W3, const float* __restrict__ b3,
    const float* __restrict__ Wo, const float* __restrict__ bo,
    float* __restrict__ out)
{
  const int b    = blockIdx.x;
  const int tid  = threadIdx.x;
  const int lane = tid & 31;
  const int wave = tid >> 5;
  const int l16  = lane & 15;
  const int hi   = lane >> 4;            // 0: lanes 0-15, 1: lanes 16-31

  extern __shared__ char smem[];
  __bf16* xbuf   = (__bf16*)(smem + OFF_X);
  __bf16* h1buf  = (__bf16*)(smem + OFF_H1);
  float*  keysF  = (float*) (smem + OFF_KEYS);
  float*  scores = (float*) (smem + OFF_SCORES);
  float*  wbar   = (float*) (smem + OFF_WBAR);
  float*  qF     = (float*) (smem + OFF_QF);
  float*  comb   = (float*) (smem + OFF_COMB);

  // ---- Phase A: keys + query -> LDS (fp32) ----
  for (int i = tid; i < SS * DD; i += 256)
    keysF[i] = keys[(size_t)b * SS * DD + i];
  if (tid < DD) qF[tid] = query[b * DD + tid];
  __syncthreads();

  // ---- Phase B: build x = [k, q, k*q, k-q] as bf16 [SP x 256] ----
  for (int i = tid; i < SP * DD; i += 256) {
    int s = i >> 6, d = i & 63;
    float kv = (s < SS) ? keysF[s * DD + d] : 0.f;
    float qv = (s < SS) ? qF[d] : 0.f;
    __bf16* row = xbuf + s * STX;
    row[d]       = (__bf16)kv;
    row[64 + d]  = (__bf16)qv;
    row[128 + d] = (__bf16)(kv * qv);
    row[192 + d] = (__bf16)(kv - qv);
  }
  __syncthreads();

  // ---- GEMM1: [SP x 256] x [256 x 256] -> h1 (PReLU, bf16) ----
  // wave w owns N-tiles {w, w+8}; B fragments held in VGPRs across M loop.
  for (int ntI = 0; ntI < 2; ++ntI) {
    const int nt   = wave + ntI * 8;
    const int head = nt >> 2;
    bf16x16 Bk[8];
#pragma unroll
    for (int kk = 0; kk < 8; ++kk)
      Bk[kk] = *(const bf16x16*)(B1pack + (((nt * 8 + kk) * 32 + lane) << 4));
    const float a1h = a1[head];
    const float b1v = b1[head * H1C + ((nt & 3) * 16 + l16)];

    for (int mt = 0; mt < MT; ++mt) {
      f32x8 acc = {};
      const __bf16* arow = xbuf + (mt * 16 + l16) * STX + hi * 8;
#pragma unroll
      for (int kk = 0; kk < 8; ++kk) {
        const __bf16* ap = arow + kk * 32;
        bf16x8 alo = *(const bf16x8*)(ap);       // K + {0..7}   (or +8..15)
        bf16x8 ahi = *(const bf16x8*)(ap + 16);  // K + {16..23} (or +24..31)
        bf16x16 av;
#pragma unroll
        for (int e = 0; e < 8; ++e) { av[e] = alo[e]; av[8 + e] = ahi[e]; }
        acc = __builtin_amdgcn_wmma_f32_16x16x32_bf16(
            false, av, false, Bk[kk], (short)0, acc, false, false);
      }
#pragma unroll
      for (int r = 0; r < 8; ++r) {              // C: lanes0-15 M=r, lanes16-31 M=r+8
        int row = mt * 16 + r + hi * 8;
        float v = acc[r] + b1v;
        v = v > 0.f ? v : a1h * v;
        h1buf[row * STX + nt * 16 + l16] = (__bf16)v;
      }
    }
  }
  __syncthreads();

  // ---- GEMM2 per head: [SP x 64] x [64 x 32] + PReLU + dot(W3) -> scores ----
  {
    const int h   = wave >> 1;
    const int mlo = (wave & 1) ? 7 : 0;
    const int mup = (wave & 1) ? MT : 7;
    bf16x16 B2[2][2];
#pragma unroll
    for (int nt2 = 0; nt2 < 2; ++nt2)
#pragma unroll
      for (int kk2 = 0; kk2 < 2; ++kk2)
        B2[nt2][kk2] = *(const bf16x16*)(B2pack +
            (((((h * 2 + nt2) * 2 + kk2) * 32 + lane)) << 4));
    const float a2h = a2[h];
    const float b3h = b3[h];
    float w3v[2], b2v[2];
#pragma unroll
    for (int nt2 = 0; nt2 < 2; ++nt2) {
      w3v[nt2] = W3[h * H2C + nt2 * 16 + l16];
      b2v[nt2] = b2[h * H2C + nt2 * 16 + l16];
    }

    for (int mt = mlo; mt < mup; ++mt) {
      f32x8 acc0 = {}, acc1 = {};
      const __bf16* arow = h1buf + (mt * 16 + l16) * STX + h * 64 + hi * 8;
#pragma unroll
      for (int kk = 0; kk < 2; ++kk) {
        const __bf16* ap = arow + kk * 32;
        bf16x8 alo = *(const bf16x8*)(ap);
        bf16x8 ahi = *(const bf16x8*)(ap + 16);
        bf16x16 av;
#pragma unroll
        for (int e = 0; e < 8; ++e) { av[e] = alo[e]; av[8 + e] = ahi[e]; }
        acc0 = __builtin_amdgcn_wmma_f32_16x16x32_bf16(
            false, av, false, B2[0][kk], (short)0, acc0, false, false);
        acc1 = __builtin_amdgcn_wmma_f32_16x16x32_bf16(
            false, av, false, B2[1][kk], (short)0, acc1, false, false);
      }
#pragma unroll
      for (int r = 0; r < 8; ++r) {
        float v0 = acc0[r] + b2v[0]; v0 = v0 > 0.f ? v0 : a2h * v0;
        float v1 = acc1[r] + b2v[1]; v1 = v1 > 0.f ? v1 : a2h * v1;
        float t  = v0 * w3v[0] + v1 * w3v[1];
        t += __shfl_xor(t, 1, 32);               // reduce over 16-lane halves
        t += __shfl_xor(t, 2, 32);
        t += __shfl_xor(t, 4, 32);
        t += __shfl_xor(t, 8, 32);
        if (l16 == 0)
          scores[h * SP + mt * 16 + r + hi * 8] = t + b3h;
      }
    }
  }
  __syncthreads();

  // ---- masked softmax over S, one wave per head ----
  if (wave < HH) {
    const int h = wave;
    const int* km = keys_mask + b * SS;
    const float NEG = -__builtin_inff();
    float sc[7];
    float mx = NEG;
#pragma unroll
    for (int it = 0; it < 7; ++it) {
      int s = lane + it * 32;
      float v = NEG;
      if (s < SS && km[s] != 0) v = scores[h * SP + s];
      sc[it] = v;
      mx = fmaxf(mx, v);
    }
#pragma unroll
    for (int o = 16; o > 0; o >>= 1) mx = fmaxf(mx, __shfl_xor(mx, o, 32));
    float sum = 0.f;
#pragma unroll
    for (int it = 0; it < 7; ++it) {
      float e = (sc[it] != NEG) ? __expf(sc[it] - mx) : 0.f;
      sc[it] = e;
      sum += e;
    }
#pragma unroll
    for (int o = 16; o > 0; o >>= 1) sum += __shfl_xor(sum, o, 32);
    float inv = (sum > 0.f) ? 1.f / sum : 0.f;   // all-masked row -> weights 0
#pragma unroll
    for (int it = 0; it < 7; ++it) {
      int s = lane + it * 32;
      if (s < SS) scores[h * SP + s] = sc[it] * inv;
    }
  }
  __syncthreads();

  // ---- mean over heads commutes with the weighted sum over keys ----
  for (int s = tid; s < SS; s += 256)
    wbar[s] = 0.25f * (scores[0 * SP + s] + scores[1 * SP + s] +
                       scores[2 * SP + s] + scores[3 * SP + s]);
  __syncthreads();

  // ---- combined = wbar @ keys (fp32), then out = combined @ Wo + bo ----
  if (tid < DD) {
    float acc = 0.f;
    for (int s = 0; s < SS; ++s) acc += wbar[s] * keysF[s * DD + tid];
    comb[tid] = acc;
  }
  __syncthreads();
  if (tid < DD) {
    float o = bo[tid];
    for (int d = 0; d < DD; ++d) o += comb[d] * Wo[d * DD + tid];
    out[b * DD + tid] = o;
  }
}

// =====================================================================
extern "C" void kernel_launch(void* const* d_in, const int* in_sizes, int n_in,
                              void* d_out, int out_size, void* d_ws, size_t ws_size,
                              hipStream_t stream) {
  const float* query = (const float*)d_in[0];
  const float* keys  = (const float*)d_in[1];
  const int*   kmask = (const int*)  d_in[2];
  const float* W1    = (const float*)d_in[3];
  const float* b1    = (const float*)d_in[4];
  const float* a1    = (const float*)d_in[5];
  const float* W2    = (const float*)d_in[6];
  const float* b2    = (const float*)d_in[7];
  const float* a2    = (const float*)d_in[8];
  const float* W3    = (const float*)d_in[9];
  const float* b3    = (const float*)d_in[10];
  const float* Wo    = (const float*)d_in[11];
  const float* bo    = (const float*)d_in[12];
  float* out = (float*)d_out;

  __bf16* B1pack = (__bf16*)d_ws;                 // 65536 bf16 = 128 KB
  __bf16* B2pack = B1pack + 16 * 8 * 32 * 16;     // 8192 bf16

  (void)hipFuncSetAttribute((const void*)rich_attn_kernel,
                            hipFuncAttributeMaxDynamicSharedMemorySize,
                            SMEM_BYTES);

  pack_weights_kernel<<<(65536 + 8192 + 255) / 256, 256, 0, stream>>>(
      W1, W2, B1pack, B2pack);
  rich_attn_kernel<<<BB, 256, SMEM_BYTES, stream>>>(
      query, keys, kmask, B1pack, B2pack,
      b1, a1, b2, a2, W3, b3, Wo, bo, out);
}
